// QuantConvNet_8211977470369
// MI455X (gfx1250) — 8-bit matrix operand order repaired and checked by reading, compiled, not run on hardware
//
#include <hip/hip_runtime.h>

typedef int   v8i __attribute__((ext_vector_type(8)));
typedef int   v2i __attribute__((ext_vector_type(2)));
typedef int   v4i __attribute__((ext_vector_type(4)));
typedef float v4f __attribute__((ext_vector_type(4)));

#define NB   32
#define CIN  64
#define HH   256
#define WW   256
#define KOUT 16
#define HWSZ (HH * WW)

#define WS_BIAS  64ull
#define WS_WPACK 128ull
#define WS_XQ    16384ull
#define WS_Y     (16384ull + 134217728ull)

__global__ void k_init(unsigned* wsu) {
    if (threadIdx.x == 0) { wsu[0] = 0u; wsu[1] = 0u; }
}

__global__ void k_absmax_x(const float4* __restrict__ x, unsigned* wsu, long n4) {
    __shared__ float red[256];
    float m = 0.f;
    long i = (long)blockIdx.x * blockDim.x + threadIdx.x;
    long stride = (long)gridDim.x * blockDim.x;
    for (; i < n4; i += stride) {
        float4 v = x[i];
        m = fmaxf(m, fmaxf(fmaxf(fabsf(v.x), fabsf(v.y)),
                           fmaxf(fabsf(v.z), fabsf(v.w))));
    }
    red[threadIdx.x] = m;
    __syncthreads();
    for (int s = 128; s > 0; s >>= 1) {
        if (threadIdx.x < s) red[threadIdx.x] = fmaxf(red[threadIdx.x], red[threadIdx.x + s]);
        __syncthreads();
    }
    if (threadIdx.x == 0) atomicMax(&wsu[0], __float_as_uint(red[0]));
}

__global__ void k_prep(const float* __restrict__ w, const float* __restrict__ b,
                       unsigned* wsu, float* wsf, int* biasq, signed char* wpack) {
    __shared__ float red[256];
    __shared__ float s_sb, s_winv;
    int t = threadIdx.x;
    float m = 0.f;
    for (int i = t; i < KOUT * CIN * 9; i += 256) m = fmaxf(m, fabsf(w[i]));
    red[t] = m;
    __syncthreads();
    for (int s = 128; s > 0; s >>= 1) {
        if (t < s) red[t] = fmaxf(red[t], red[t + s]);
        __syncthreads();
    }
    if (t == 0) {
        float wmax = red[0];
        float xmax = __uint_as_float(wsu[0]);
        float sin_ = xmax / 127.f, sw_ = wmax / 7.f, sb_ = sin_ * sw_;
        wsf[2] = sin_; wsf[3] = sw_; wsf[4] = sb_;
        wsf[5] = (xmax > 0.f) ? (127.f / xmax) : 0.f;
        wsf[6] = (wmax > 0.f) ? (7.f / wmax) : 0.f;
        s_sb = sb_; s_winv = wsf[6];
    }
    __syncthreads();
    if (t < KOUT) {
        float q = rintf(b[t] / s_sb);
        q = fminf(fmaxf(q, -2147483648.f), 2147483520.f);
        biasq[t] = (int)q;
    }
    for (int idx = t; idx < 9 * 32 * 32; idx += 256) {
        int tap = idx >> 10, rem = idx & 1023, lane = rem >> 5, byte = rem & 31;
        int v = byte >> 2, bb = byte & 3;
        int n = lane & 15, khalf = (lane >> 4) * 8;
        int k = ((byte >> 3) * 16) + khalf + (byte & 7);
        float q = rintf(w[(n * CIN + k) * 9 + tap] * s_winv);
        q = fminf(fmaxf(q, -8.f), 7.f);
        wpack[tap * 1024 + lane * 32 + byte] = (signed char)q;
    }
}

__global__ void k_quant_x(const float* __restrict__ x, const float* __restrict__ wsf,
                          signed char* __restrict__ xq) {
    __shared__ __align__(16) signed char lds[64 * 80];
    float invs = wsf[5];
    int t = threadIdx.x;
    int blk = blockIdx.x;
    int wblk = blk & 3, h = (blk >> 2) & 255, n = blk >> 10;
    int w = t & 63;
    int w0 = wblk * 64;
#pragma unroll
    for (int i = 0; i < 16; i++) {
        int c = (t >> 6) + i * 4;
        float v = x[((size_t)(n * CIN + c)) * HWSZ + (size_t)h * WW + w0 + w];
        float q = rintf(v * invs);
        q = fminf(fmaxf(q, -128.f), 127.f);
        lds[w * 80 + c] = (signed char)q;
    }
    __syncthreads();
    int p = t >> 2, q4 = t & 3;
    v4i val = *(const v4i*)(&lds[p * 80 + q4 * 16]);
    *(v4i*)(&xq[(((size_t)(n * HH + h)) * WW + w0 + p) * 64 + (size_t)q4 * 16]) = val;
}

__global__ void k_conv(const signed char* __restrict__ xq,
                       const signed char* __restrict__ wpack,
                       const int* __restrict__ biasq,
                       int* __restrict__ y, unsigned* wsu) {
    __shared__ int redi[256];
    int t = threadIdx.x;
    int lane = t & 31;
    int waveId = blockIdx.x * 8 + (t >> 5);
    int wt = waveId & 15, h = (waveId >> 4) & 255, n = waveId >> 12;
    int m = lane & 15;
    int half = (lane >> 4) * 8;
    const signed char* abase =
        xq + (((size_t)(n * HH + h)) * WW + (size_t)(wt * 16 + m)) * 64 + half;

    v8i afrag[9];
    bool interior = ((unsigned)(h - 1) < 254u) && ((unsigned)(wt - 1) < 14u);
    if (interior) {
#pragma unroll
        for (int tap = 0; tap < 9; tap++) {
            const int r = tap / 3, s = tap % 3;
            const signed char* p = abase + ((r - 1) * WW + (s - 1)) * 64;
            v2i x0 = *(const v2i*)(p);
            v2i x1 = *(const v2i*)(p + 16);
            v2i x2 = *(const v2i*)(p + 32);
            v2i x3 = *(const v2i*)(p + 48);
            afrag[tap] = (v8i){x0.x, x0.y, x1.x, x1.y, x2.x, x2.y, x3.x, x3.y};
        }
    } else {
#pragma unroll
        for (int tap = 0; tap < 9; tap++) {
            const int r = tap / 3, s = tap % 3;
            int hin = h + r - 1;
            int win = wt * 16 + m + s - 1;
            v8i a = {0, 0, 0, 0, 0, 0, 0, 0};
            if ((unsigned)hin < 256u && (unsigned)win < 256u) {
                const signed char* p = abase + ((r - 1) * WW + (s - 1)) * 64;
                v2i x0 = *(const v2i*)(p);
                v2i x1 = *(const v2i*)(p + 16);
                v2i x2 = *(const v2i*)(p + 32);
                v2i x3 = *(const v2i*)(p + 48);
                a = (v8i){x0.x, x0.y, x1.x, x1.y, x2.x, x2.y, x3.x, x3.y};
            }
            afrag[tap] = a;
        }
    }

    v8i acc = {0, 0, 0, 0, 0, 0, 0, 0};
#pragma unroll
    for (int tap = 0; tap < 9; tap++) {
        v8i bm = *(const v8i*)(wpack + tap * 1024 + lane * 32);
        acc = __builtin_amdgcn_wmma_i32_16x16x64_iu8(true, afrag[tap], true, bm, acc,
                                                     false, false);
    }

    int nch = lane & 15;
    int bq = biasq[nch];
    size_t base = ((size_t)(n * KOUT + nch)) * HWSZ + (size_t)h * WW
                + (size_t)(wt * 16 + half);
    v4i s0 = {acc[0] + bq, acc[1] + bq, acc[2] + bq, acc[3] + bq};
    v4i s1 = {acc[4] + bq, acc[5] + bq, acc[6] + bq, acc[7] + bq};
    __builtin_nontemporal_store(s0, (v4i*)(&y[base]));
    __builtin_nontemporal_store(s1, (v4i*)(&y[base + 4]));
    int mymax = 0;
    {
        int a0 = abs(s0.x), a1 = abs(s0.y), a2 = abs(s0.z), a3 = abs(s0.w);
        int a4 = abs(s1.x), a5 = abs(s1.y), a6 = abs(s1.z), a7 = abs(s1.w);
        mymax = max(max(max(a0, a1), max(a2, a3)), max(max(a4, a5), max(a6, a7)));
    }
    redi[t] = mymax;
    __syncthreads();
    for (int s2 = 128; s2 > 0; s2 >>= 1) {
        if (t < s2) redi[t] = redi[t] > redi[t + s2] ? redi[t] : redi[t + s2];
        __syncthreads();
    }
    if (t == 0) atomicMax(&wsu[1], (unsigned)redi[0]);
}

__global__ void k_requant(const v4i* __restrict__ y, const unsigned* __restrict__ wsu,
                          const float* __restrict__ wsf, v4f* __restrict__ out, long n4) {
    float sb = wsf[4];
    unsigned im = wsu[1];
    float imf = (float)im;
    float inv = (im > 0u) ? (127.f / imf) : 0.f;
    float sout = sb * imf / 127.f;
    long i = (long)blockIdx.x * blockDim.x + threadIdx.x;
    long stride = (long)gridDim.x * blockDim.x;
    for (; i < n4; i += stride) {
        v4i v = y[i];
        v4f o;
        o.x = fminf(fmaxf(rintf((float)v.x * inv), -128.f), 127.f) * sout;
        o.y = fminf(fmaxf(rintf((float)v.y * inv), -128.f), 127.f) * sout;
        o.z = fminf(fmaxf(rintf((float)v.z * inv), -128.f), 127.f) * sout;
        o.w = fminf(fmaxf(rintf((float)v.w * inv), -128.f), 127.f) * sout;
        __builtin_nontemporal_store(o, &out[i]);
    }
}

extern "C" void kernel_launch(void* const* d_in, const int* in_sizes, int n_in,
                              void* d_out, int out_size, void* d_ws, size_t ws_size,
                              hipStream_t stream) {
    const float* x = (const float*)d_in[0];
    const float* w = (const float*)d_in[1];
    const float* b = (const float*)d_in[2];
    unsigned* wsu = (unsigned*)d_ws;
    float* wsf = (float*)d_ws;
    int* biasq = (int*)((char*)d_ws + WS_BIAS);
    signed char* wpack = (signed char*)((char*)d_ws + WS_WPACK);
    signed char* xq = (signed char*)((char*)d_ws + WS_XQ);
    int* y = (int*)((char*)d_ws + WS_Y);
    float* out = (float*)d_out;

    k_init<<<1, 32, 0, stream>>>(wsu);
    long n4 = (long)NB * CIN * HWSZ / 4;
    k_absmax_x<<<4096, 256, 0, stream>>>((const float4*)x, wsu, n4);
    k_prep<<<1, 256, 0, stream>>>(w, b, wsu, wsf, biasq, wpack);
    k_quant_x<<<NB * HH * 4, 256, 0, stream>>>(x, wsf, xq);
    k_conv<<<NB * HH * 16 / 8, 256, 0, stream>>>(xq, wpack, biasq, y, wsu);
    long ny4 = (long)NB * KOUT * HWSZ / 4;
    k_requant<<<8192, 256, 0, stream>>>((const v4i*)y, wsu, wsf, (v4f*)out, ny4);
}
